// AttentionBlock_59090160058779
// MI455X (gfx1250) — matrix operand order repaired and checked by reading, not compiled, not run on hardware
//
#include <hip/hip_runtime.h>
#include <hip/hip_bf16.h>
#include <stdint.h>

typedef __attribute__((ext_vector_type(16))) __bf16 v16bf;
typedef __attribute__((ext_vector_type(8)))  float  v8f;

#define N_SEQ 4096
#define DIM   512
#define KTILE 64            // K rows per iteration tile
#define WAVES 8
#define BLOCK (WAVES * 32)
#define QROWS (WAVES * 16)  // 128 query rows per workgroup

// LDS layout (bytes)
#define KR_OFF 0            // bf16 [64][512]   = 65536  (row-major K tile)
#define KT_OFF 65536        // bf16 [512][64]   = 65536  (transposed K tile)
#define QS_OFF 131072       // bf16 [128][512]  = 131072 (Q tile)
#define PS_OFF 262144       // bf16 [8][16][64] = 16384  (per-wave P staging)
#define SMEM_BYTES 278528   // 272 KB of the WGP's 320 KB LDS

__device__ __forceinline__ uint32_t bf16rne(float f) {
    uint32_t u = __float_as_uint(f);
    return (u + 0x7FFFu + ((u >> 16) & 1u)) >> 16;
}
__device__ __forceinline__ uint32_t pack2(float a, float b) {
    return bf16rne(a) | (bf16rne(b) << 16);
}

union FragAB { v16bf v; uint4 q[2]; };

__global__ void __launch_bounds__(BLOCK, 1)
flash_attn_xxt_kernel(const float* __restrict__ x, float* __restrict__ out)
{
    extern __shared__ char smem[];
    char* kr = smem + KR_OFF;
    char* kt = smem + KT_OFF;
    char* qs = smem + QS_OFF;
    char* ps = smem + PS_OFF;

    const int b     = blockIdx.y;
    const int qbase = blockIdx.x * QROWS;
    const int tid   = threadIdx.x;
    const int wave  = tid >> 5;
    const int lane  = tid & 31;
    const int l16   = lane & 15;
    const int hi    = lane >> 4;
    const int k0    = hi * 8;    // A-frag K sub-block base
    const int kb2   = hi * 16;   // unused: B-frags use k0 too

    const float* xb = x + (size_t)b * N_SEQ * DIM;

    // ---- stage Q tile (rows qbase..qbase+127) -> qs as bf16 row-major ----
    for (int t = tid; t < QROWS * 128; t += BLOCK) {
        const int r  = t >> 7;
        const int c4 = t & 127;
        const float4 f = *(const float4*)(xb + (size_t)(qbase + r) * DIM + c4 * 4);
        uint2 p; p.x = pack2(f.x, f.y); p.y = pack2(f.z, f.w);
        *(uint2*)(qs + (size_t)(r * DIM + c4 * 4) * 2) = p;
    }
    __syncthreads();

    // per-lane row stats: lane owns rows v + 8*hi (v = 0..7) of its wave's 16
    float m[8], l[8];
    v8f o[32];
    #pragma unroll
    for (int v = 0; v < 8; ++v) { m[v] = -3.0e38f; l[v] = 0.0f; }
    #pragma unroll
    for (int t = 0; t < 32; ++t)
        #pragma unroll
        for (int v = 0; v < 8; ++v) o[t][v] = 0.0f;

    for (int kb = 0; kb < N_SEQ; kb += KTILE) {
        // ---- stage K tile: row-major bf16 (kr) + transposed bf16 (kt) ----
        for (int t = tid; t < 32 * 128; t += BLOCK) {
            const int p2 = t >> 7;     // row pair 0..31
            const int c4 = t & 127;    // float4 column index
            const float* r0 = xb + (size_t)(kb + 2 * p2) * DIM + c4 * 4;
            const float4 f0 = *(const float4*)(r0);
            const float4 f1 = *(const float4*)(r0 + DIM);
            uint2 a; a.x = pack2(f0.x, f0.y); a.y = pack2(f0.z, f0.w);
            uint2 c; c.x = pack2(f1.x, f1.y); c.y = pack2(f1.z, f1.w);
            *(uint2*)(kr + (size_t)((2 * p2)     * DIM + c4 * 4) * 2) = a;
            *(uint2*)(kr + (size_t)((2 * p2 + 1) * DIM + c4 * 4) * 2) = c;
            char* kt0 = kt + (size_t)((c4 * 4) * KTILE + 2 * p2) * 2;
            *(uint32_t*)(kt0)                 = pack2(f0.x, f1.x);
            *(uint32_t*)(kt0 + 1 * KTILE * 2) = pack2(f0.y, f1.y);
            *(uint32_t*)(kt0 + 2 * KTILE * 2) = pack2(f0.z, f1.z);
            *(uint32_t*)(kt0 + 3 * KTILE * 2) = pack2(f0.w, f1.w);
        }
        if (kb + KTILE < N_SEQ)  // nudge next tile toward L2
            __builtin_prefetch(xb + (size_t)(kb + KTILE) * DIM + tid * 32, 0, 1);
        __syncthreads();

        // ---- GEMM1: S(16x64) = Q(16x512) * Ktile^T, bf16 WMMA f32-acc ----
        v8f s[4];
        #pragma unroll
        for (int c = 0; c < 4; ++c)
            #pragma unroll
            for (int v = 0; v < 8; ++v) s[c][v] = 0.0f;

        const int qrow = wave * 16 + l16;
        #pragma unroll
        for (int f = 0; f < DIM / 32; ++f) {
            FragAB a;
            const char* ab = qs + (size_t)(qrow * DIM + f * 32 + k0) * 2;
            a.q[0] = *(const uint4*)(ab);
            a.q[1] = *(const uint4*)(ab + 32);
            FragAB bb[4];                       // all 4 column tiles live at once
            #pragma unroll
            for (int c = 0; c < 4; ++c) {
                const char* bp = kr + (size_t)((c * 16 + l16) * DIM + f * 32 + k0) * 2;
                bb[c].q[0] = *(const uint4*)(bp);
                bb[c].q[1] = *(const uint4*)(bp + 32);
            }
            #pragma unroll
            for (int c = 0; c < 4; ++c)
                s[c] = __builtin_amdgcn_wmma_f32_16x16x32_bf16(
                    false, a.v, false, bb[c].v, (short)0, s[c], false, false);
        }

        // ---- online softmax update ----
        float alpha[8];
        #pragma unroll
        for (int v = 0; v < 8; ++v) {
            float rv = fmaxf(fmaxf(s[0][v], s[1][v]), fmaxf(s[2][v], s[3][v]));
            rv = fmaxf(rv, __shfl_xor(rv, 1, 32));
            rv = fmaxf(rv, __shfl_xor(rv, 2, 32));
            rv = fmaxf(rv, __shfl_xor(rv, 4, 32));
            rv = fmaxf(rv, __shfl_xor(rv, 8, 32));
            const float mn = fmaxf(m[v], rv);
            alpha[v] = __expf(m[v] - mn);
            m[v] = mn;
        }
        #pragma unroll
        for (int v = 0; v < 8; ++v) {
            float rs = 0.0f;
            #pragma unroll
            for (int c = 0; c < 4; ++c) {
                const float e = __expf(s[c][v] - m[v]);
                s[c][v] = e;
                rs += e;
            }
            rs += __shfl_xor(rs, 1, 32);
            rs += __shfl_xor(rs, 2, 32);
            rs += __shfl_xor(rs, 4, 32);
            rs += __shfl_xor(rs, 8, 32);
            l[v] = l[v] * alpha[v] + rs;
        }
        #pragma unroll
        for (int t = 0; t < 32; ++t)
            #pragma unroll
            for (int v = 0; v < 8; ++v) o[t][v] *= alpha[v];

        // ---- P -> per-wave LDS staging (C-layout -> A-layout round trip) ----
        {
            unsigned short* pw = (unsigned short*)(ps + wave * (16 * 64 * 2));
            #pragma unroll
            for (int c = 0; c < 4; ++c)
                #pragma unroll
                for (int v = 0; v < 8; ++v)
                    pw[(v + 8 * hi) * 64 + c * 16 + l16] =
                        (unsigned short)bf16rne(s[c][v]);
        }
        asm volatile("s_wait_dscnt 0x0" ::: "memory");  // same-wave LDS RAW

        // ---- GEMM2: O(16x512) += P(16x64) * Ktile(64x512) ----
        #pragma unroll
        for (int ss = 0; ss < 2; ++ss) {
            FragAB a;
            const char* ab = ps + wave * (16 * 64 * 2)
                           + (size_t)(l16 * 64 + ss * 32 + k0) * 2;
            a.q[0] = *(const uint4*)(ab);
            a.q[1] = *(const uint4*)(ab + 32);
            #pragma unroll
            for (int tg = 0; tg < 8; ++tg) {     // groups of 4 d-tiles
                FragAB bb[4];
                #pragma unroll
                for (int j = 0; j < 4; ++j) {
                    const int t = tg * 4 + j;
                    const char* bp = kt + (size_t)((t * 16 + l16) * KTILE + ss * 32 + k0) * 2;
                    bb[j].q[0] = *(const uint4*)(bp);
                    bb[j].q[1] = *(const uint4*)(bp + 32);
                }
                #pragma unroll
                for (int j = 0; j < 4; ++j)
                    o[tg * 4 + j] = __builtin_amdgcn_wmma_f32_16x16x32_bf16(
                        false, a.v, false, bb[j].v, (short)0, o[tg * 4 + j], false, false);
            }
        }
        __syncthreads();   // before next iteration overwrites kr/kt
    }

    // ---- epilogue: O /= l, store fp32 ----
    float* ob = out + (size_t)b * N_SEQ * DIM;
    float inv[8];
    #pragma unroll
    for (int v = 0; v < 8; ++v) inv[v] = 1.0f / l[v];
    const int rowbase = qbase + wave * 16 + 8 * hi;
    #pragma unroll
    for (int t = 0; t < 32; ++t)
        #pragma unroll
        for (int v = 0; v < 8; ++v)
            ob[(size_t)(rowbase + v) * DIM + t * 16 + l16] = o[t][v] * inv[v];
}

extern "C" void kernel_launch(void* const* d_in, const int* in_sizes, int n_in,
                              void* d_out, int out_size, void* d_ws, size_t ws_size,
                              hipStream_t stream) {
    (void)in_sizes; (void)n_in; (void)out_size; (void)d_ws; (void)ws_size;
    const float* x = (const float*)d_in[0];
    float* out = (float*)d_out;

    hipFuncSetAttribute((const void*)flash_attn_xxt_kernel,
                        hipFuncAttributeMaxDynamicSharedMemorySize, SMEM_BYTES);

    dim3 grid(N_SEQ / QROWS, 8);   // 32 x 8 = 256 workgroups
    dim3 block(BLOCK);             // 8 wave32s
    flash_attn_xxt_kernel<<<grid, block, SMEM_BYTES, stream>>>(x, out);
}
